// BaselineFraudDetector_90804198572577
// MI455X (gfx1250) — matrix operand order repaired and checked by reading, not compiled, not run on hardware
//
#include <hip/hip_runtime.h>
#include <stdint.h>

#define T_TYPES 11
#define NUM_TX  131072
#define N_ENT   131072
#define E_EDGES 262144
#define H_DIM   128
#define TXF     394
#define LN_EPS  1e-5f

// ---------------------------------------------------------------- WMMA types
typedef __attribute__((ext_vector_type(16))) __bf16 v16bf;
typedef __attribute__((ext_vector_type(2)))  __bf16 v2bf;
typedef __attribute__((ext_vector_type(8)))  float  v8f;
typedef uint32_t u32x4 __attribute__((ext_vector_type(4)));
typedef uint32_t u32x8 __attribute__((ext_vector_type(8)));

union Frag16 { v16bf v; uint32_t u[8]; };
union FragC  { v8f   v; float    f[8]; };

__device__ __forceinline__ uint32_t pack2_bf16(float a, float b) {
#if __has_builtin(__builtin_amdgcn_cvt_pk_bf16_f32)
  v2bf p = __builtin_amdgcn_cvt_pk_bf16_f32(a, b);
  return __builtin_bit_cast(uint32_t, p);
#else
  union { uint32_t u; __bf16 h[2]; } r;
  r.h[0] = (__bf16)a; r.h[1] = (__bf16)b;
  return r.u;
#endif
}

__device__ __forceinline__ v8f wmma_bf16(const Frag16& a, const Frag16& b, v8f c) {
  return __builtin_amdgcn_wmma_f32_16x16x32_bf16(false, a.v, false, b.v,
                                                 (short)0, c, false, false);
}

// Build a 16x32 bf16 A fragment from a row-major f32 row pointer.
// ISA layout: lanes 0-15 row M=lane (K 0..7 in u[0..3], K 16..23 in u[4..7]);
// lanes 16-31 same rows (K 8..15 / K 24..31).
__device__ __forceinline__ void build_afrag(Frag16& a, const float* __restrict__ row,
                                            int kbase, int krows, int lane) {
  int half = lane >> 4;
  int c0 = kbase + half * 8;
#pragma unroll
  for (int j = 0; j < 4; ++j) {
    int c = c0 + 2 * j;
    float f0 = (c     < krows) ? row[c]     : 0.f;
    float f1 = (c + 1 < krows) ? row[c + 1] : 0.f;
    a.u[j] = pack2_bf16(f0, f1);
  }
  int c1 = c0 + 16;
#pragma unroll
  for (int j = 0; j < 4; ++j) {
    int c = c1 + 2 * j;
    float f0 = (c     < krows) ? row[c]     : 0.f;
    float f1 = (c + 1 < krows) ? row[c + 1] : 0.f;
    a.u[4 + j] = pack2_bf16(f0, f1);
  }
}

// ordered-uint float key for atomicMax-based segment max
__device__ __forceinline__ uint32_t fkey(float f) {
  uint32_t u = __float_as_uint(f);
  return (u & 0x80000000u) ? ~u : (u | 0x80000000u);
}
__device__ __forceinline__ float fdekey(uint32_t k) {
  uint32_t u = (k & 0x80000000u) ? (k & 0x7fffffffu) : ~k;
  return __uint_as_float(u);
}

// --------------------------------------------------------------- TDM helper
// 1-D global->LDS DMA of `nwords` dwords (wave-uniform), ISA cdna5 §8 D# layout.
__device__ __forceinline__ void tdm_load_1d(const uint32_t* gsrc, uint32_t lds_byte_addr,
                                            uint32_t nwords) {
  uint64_t ga = (uint64_t)(uintptr_t)gsrc;
  u32x4 g0;
  g0.x = 1u;                                    // count=1, user descriptor
  g0.y = lds_byte_addr;                         // lds_addr [63:32]
  g0.z = (uint32_t)ga;                          // global_addr lo
  g0.w = ((uint32_t)(ga >> 32) & 0x01ffffffu)   // global_addr [56:32]
         | (2u << 30);                          // type=2 (image)
  u32x8 g1;
  g1[0] = 0x00020000u;                          // wg_mask=0, data_size=2 (4B)
  g1[1] = (nwords & 0xffffu) << 16;             // tensor_dim0 lo16 at bits[63:48]
  g1[2] = (nwords >> 16) | (1u << 16);          // tensor_dim0 hi16, tensor_dim1=1
  g1[3] = (nwords & 0xffffu) << 16;             // tile_dim0 at bits[127:112]
  g1[4] = 1u;                                   // tile_dim1=1, tile_dim2=0
  g1[5] = nwords;                               // tensor_dim0_stride lo32
  g1[6] = 0u;                                   // stride hi16, dim1_stride lo16
  g1[7] = 0u;
  asm volatile("tensor_load_to_lds %0, %1" :: "s"(g0), "s"(g1) : "memory");
}

// ------------------------------------------------------------- weight packer
// kt-major fragment order: f = kt*Nt + nt  (so one K-step is contiguous for TDM).
// Per lane 8 dwords; lane&15 = column within N-tile, lane>>4 = K half;
// u[j] = bf16 pair (K = koff+2(j&3)+16(j>>2), +1), koff = kt*32 + (lane>>4)*8.
__global__ void pack_b_kernel(const float* __restrict__ W, int Krows, int Kt, int Nt,
                              int Ncols, uint32_t* __restrict__ out) {
  int tid = blockIdx.x * blockDim.x + threadIdx.x;
  int total = Kt * Nt * 32;
  if (tid >= total) return;
  int lane = tid & 31;
  int f = tid >> 5;
  int kt = f / Nt, nt = f % Nt;
  int n = nt * 16 + (lane & 15);
  int koff = kt * 32 + (lane >> 4) * 8;
  uint32_t* dst = out + (size_t)(f * 32 + lane) * 8;
#pragma unroll
  for (int j = 0; j < 8; ++j) {
    int k0 = koff + 2 * (j & 3) + (j >> 2) * 16;
    float a = (k0     < Krows) ? W[(size_t)k0       * Ncols + n] : 0.f;
    float b = (k0 + 1 < Krows) ? W[(size_t)(k0 + 1) * Ncols + n] : 0.f;
    dst[j] = pack2_bf16(a, b);
  }
}

// --------------------------------------------------- fused gather + QKVS GEMM
// block = 128 threads (4 waves); A-tile fragments built cooperatively (wave w
// builds K-tile w into LDS), then wave w computes x@W_w + b_w for the 16x128 tile.
__global__ void __launch_bounds__(128)
qkvs_kernel(const float* __restrict__ emb, const int* __restrict__ idx,
            const uint32_t* __restrict__ packedW,
            const float* __restrict__ bq, const float* __restrict__ bk,
            const float* __restrict__ bv, const float* __restrict__ bs,
            float* __restrict__ q, float* __restrict__ k,
            float* __restrict__ v, float* __restrict__ ss) {
  __shared__ uint32_t ldsA[4 * 256];
  int wave = threadIdx.x >> 5;
  int lane = threadIdx.x & 31;
  int m0 = blockIdx.x * 16;
  int r = lane & 15, half = lane >> 4;

  int g = idx[m0 + r];
  const float* xrow = emb + (size_t)g * H_DIM;
  {
    Frag16 tmp;
    build_afrag(tmp, xrow, wave * 32, H_DIM, lane);   // wave builds K-tile = wave
#pragma unroll
    for (int j = 0; j < 8; ++j) ldsA[wave * 256 + lane * 8 + j] = tmp.u[j];
  }
  __syncthreads();
  Frag16 afrag[4];
#pragma unroll
  for (int kt = 0; kt < 4; ++kt)
#pragma unroll
    for (int j = 0; j < 8; ++j) afrag[kt].u[j] = ldsA[kt * 256 + lane * 8 + j];

  const float* bias = (wave == 0) ? bq : (wave == 1) ? bk : (wave == 2) ? bv : bs;
  float* out       = (wave == 0) ? q  : (wave == 1) ? k  : (wave == 2) ? v  : ss;
  const uint32_t* pB = packedW + (size_t)wave * (4 * 8 * 32 * 8);

#pragma unroll
  for (int nt = 0; nt < 8; ++nt) {
    FragC acc;
    float bval = bias[nt * 16 + r];
#pragma unroll
    for (int i = 0; i < 8; ++i) acc.f[i] = bval;
#pragma unroll
    for (int kt = 0; kt < 4; ++kt) {
      Frag16 bfrag;
      const uint32_t* p = pB + (size_t)((kt * 8 + nt) * 32 + lane) * 8;
#pragma unroll
      for (int j = 0; j < 8; ++j) bfrag.u[j] = p[j];
      acc.v = wmma_bf16(afrag[kt], bfrag, acc.v);
    }
#pragma unroll
    for (int i = 0; i < 8; ++i)
      out[(size_t)(m0 + i + 8 * half) * H_DIM + nt * 16 + r] = acc.f[i];
  }
}

// ----------------------------------------------------------- generic WMMA GEMM
// C[M x (NT*16)] (+)= A[M x krows] @ packedB ; one 16-row tile per wave,
// 8 waves / block. B fragments DMA'd global->LDS by the Tensor Data Mover,
// double-buffered (issue kt+1, wait tensorcnt<=1, compute kt).
// mode: 0 = bias-init + store, 1 = accumulate into C, 2 = bias-init + relu store
template <int NT>
__global__ void __launch_bounds__(256)
gemm_wmma_kernel(const float* __restrict__ A, int lda, int krows, int ktiles,
                 const uint32_t* __restrict__ packedB,
                 const float* __restrict__ bias, float* __restrict__ C, int ldc,
                 int mode) {
  __shared__ uint32_t ldsB[2][NT * 256];
  int wave = threadIdx.x >> 5, lane = threadIdx.x & 31;
  int tile = blockIdx.x * 8 + wave;
  int m0 = tile * 16;
  int r = lane & 15, half = lane >> 4;
  const float* arow = A + (size_t)(m0 + r) * lda;

  FragC acc[NT];
#pragma unroll
  for (int nt = 0; nt < NT; ++nt) {
    if (mode == 1) {
#pragma unroll
      for (int i = 0; i < 8; ++i)
        acc[nt].f[i] = C[(size_t)(m0 + i + 8 * half) * ldc + nt * 16 + r];
    } else {
      float bval = bias[nt * 16 + r];
#pragma unroll
      for (int i = 0; i < 8; ++i) acc[nt].f[i] = bval;
    }
  }

  if (wave == 0)
    tdm_load_1d(packedB, (uint32_t)(uintptr_t)&ldsB[0][0], NT * 256);

  for (int kt = 0; kt < ktiles; ++kt) {
    __syncthreads();            // everyone done reading buffer (kt+1)&1
    if (wave == 0) {
      if (kt + 1 < ktiles) {
        tdm_load_1d(packedB + (size_t)(kt + 1) * NT * 256,
                    (uint32_t)(uintptr_t)&ldsB[(kt + 1) & 1][0], NT * 256);
        __builtin_amdgcn_s_wait_tensorcnt(1);   // retire DMA for kt
      } else {
        __builtin_amdgcn_s_wait_tensorcnt(0);
      }
    }
    __syncthreads();            // publish buffer kt&1
    Frag16 afrag;
    build_afrag(afrag, arow, kt * 32, krows, lane);
    const uint32_t* bp = &ldsB[kt & 1][0];
#pragma unroll
    for (int nt = 0; nt < NT; ++nt) {
      Frag16 bfrag;
#pragma unroll
      for (int j = 0; j < 8; ++j) bfrag.u[j] = bp[nt * 256 + lane * 8 + j];
      acc[nt].v = wmma_bf16(afrag, bfrag, acc[nt].v);
    }
  }

#pragma unroll
  for (int nt = 0; nt < NT; ++nt) {
#pragma unroll
    for (int i = 0; i < 8; ++i) {
      float vv = acc[nt].f[i];
      if (mode == 2) vv = fmaxf(vv, 0.f);
      C[(size_t)(m0 + i + 8 * half) * ldc + nt * 16 + r] = vv;
    }
  }
}

// ----------------------------------------------------------- edge attention
__global__ void __launch_bounds__(256)
score_kernel(const float* __restrict__ q, const float* __restrict__ k,
             const int* __restrict__ src, const int* __restrict__ dst,
             float* __restrict__ sc, uint32_t* __restrict__ mkey, float scale) {
  int wave = threadIdx.x >> 5, lane = threadIdx.x & 31;
  int e = blockIdx.x * 8 + wave;
  int s = src[e], d = dst[e];
  float4 a = ((const float4*)(q + (size_t)d * H_DIM))[lane];
  float4 b = ((const float4*)(k + (size_t)s * H_DIM))[lane];
  float p = a.x * b.x + a.y * b.y + a.z * b.z + a.w * b.w;
#pragma unroll
  for (int off = 16; off; off >>= 1) p += __shfl_xor(p, off, 32);
  if (lane == 0) {
    float val = p * scale;
    sc[e] = val;
    atomicMax(&mkey[d], fkey(val));
  }
}

__global__ void __launch_bounds__(256)
expsum_kernel(const float* __restrict__ sc, const int* __restrict__ dst,
              const uint32_t* __restrict__ mkey,
              float* __restrict__ ew, float* __restrict__ den) {
  int e = blockIdx.x * blockDim.x + threadIdx.x;
  int d = dst[e];
  float x = __expf(sc[e] - fdekey(mkey[d]));
  ew[e] = x;
  atomicAdd(&den[d], x);
}

__global__ void __launch_bounds__(256)
msg_kernel(const float* __restrict__ v, const float* __restrict__ ew,
           const float* __restrict__ den, const int* __restrict__ src,
           const int* __restrict__ dst, float* __restrict__ msg) {
  int wave = threadIdx.x >> 5, lane = threadIdx.x & 31;
  int e = blockIdx.x * 8 + wave;
  int s = src[e], d = dst[e];
  float alpha = ew[e] / den[d];
  float4 val = ((const float4*)(v + (size_t)s * H_DIM))[lane];
  float* mr = msg + (size_t)d * H_DIM + lane * 4;
  atomicAdd(mr + 0, alpha * val.x);
  atomicAdd(mr + 1, alpha * val.y);
  atomicAdd(mr + 2, alpha * val.z);
  atomicAdd(mr + 3, alpha * val.w);
}

// h = LayerNorm(msg + skip) * g + b, in place over msg (one wave per row)
__global__ void __launch_bounds__(256)
norm_kernel(float* __restrict__ msg, const float* __restrict__ ss,
            const float* __restrict__ g, const float* __restrict__ bln) {
  int wave = threadIdx.x >> 5, lane = threadIdx.x & 31;
  int row = blockIdx.x * 8 + wave;
  float4 mv = ((const float4*)(msg + (size_t)row * H_DIM))[lane];
  float4 sv = ((const float4*)(ss  + (size_t)row * H_DIM))[lane];
  float4 h;
  h.x = mv.x + sv.x; h.y = mv.y + sv.y; h.z = mv.z + sv.z; h.w = mv.w + sv.w;
  float s1 = h.x + h.y + h.z + h.w;
  float s2 = h.x * h.x + h.y * h.y + h.z * h.z + h.w * h.w;
#pragma unroll
  for (int off = 16; off; off >>= 1) {
    s1 += __shfl_xor(s1, off, 32);
    s2 += __shfl_xor(s2, off, 32);
  }
  float mu = s1 * (1.f / 128.f);
  float var = s2 * (1.f / 128.f) - mu * mu;
  float rstd = rsqrtf(var + LN_EPS);
  int c = lane * 4;
  float4 o;
  o.x = (h.x - mu) * rstd * g[c + 0] + bln[c + 0];
  o.y = (h.y - mu) * rstd * g[c + 1] + bln[c + 1];
  o.z = (h.z - mu) * rstd * g[c + 2] + bln[c + 2];
  o.w = (h.w - mu) * rstd * g[c + 3] + bln[c + 3];
  ((float4*)(msg + (size_t)row * H_DIM))[lane] = o;
}

__global__ void __launch_bounds__(256)
scatter_kernel(const float* __restrict__ h, const int* __restrict__ src,
               const int* __restrict__ dst, float* __restrict__ msgsb) {
  int wave = threadIdx.x >> 5, lane = threadIdx.x & 31;
  int e = blockIdx.x * 8 + wave;
  int s = src[e], d = dst[e];
  float4 val = ((const float4*)(h + (size_t)s * H_DIM))[lane];
  float* mr = msgsb + (size_t)d * H_DIM + lane * 4;
  atomicAdd(mr + 0, val.x);
  atomicAdd(mr + 1, val.y);
  atomicAdd(mr + 2, val.z);
  atomicAdd(mr + 3, val.w);
}

__global__ void __launch_bounds__(256) relu_kernel(float4* __restrict__ x) {
  int i = blockIdx.x * blockDim.x + threadIdx.x;
  float4 v = x[i];
  v.x = fmaxf(v.x, 0.f); v.y = fmaxf(v.y, 0.f);
  v.z = fmaxf(v.z, 0.f); v.w = fmaxf(v.w, 0.f);
  x[i] = v;
}

__global__ void __launch_bounds__(256)
out_kernel(const float* __restrict__ h2, const float* __restrict__ W3,
           const float* __restrict__ b3, float* __restrict__ out) {
  int wave = threadIdx.x >> 5, lane = threadIdx.x & 31;
  int row = blockIdx.x * 8 + wave;
  const float* hr = h2 + (size_t)row * 64;
  float p = hr[2 * lane] * W3[2 * lane] + hr[2 * lane + 1] * W3[2 * lane + 1];
#pragma unroll
  for (int off = 16; off; off >>= 1) p += __shfl_xor(p, off, 32);
  if (lane == 0) out[row] = p + b3[0];
}

// --------------------------------------------------------------------- launch
extern "C" void kernel_launch(void* const* d_in, const int* in_sizes, int n_in,
                              void* d_out, int out_size, void* d_ws, size_t ws_size,
                              hipStream_t stream) {
  (void)in_sizes; (void)n_in; (void)out_size; (void)ws_size;
  const float* tx_x = (const float*)d_in[0];
  const float* emb  = (const float*)d_in[1];
  const int*   eidx = (const int*)d_in[2];
  const int*   esrc = (const int*)d_in[3];
  const int*   edst = (const int*)d_in[4];
  const float* Wq = (const float*)d_in[5];  const float* bq = (const float*)d_in[6];
  const float* Wk = (const float*)d_in[7];  const float* bk = (const float*)d_in[8];
  const float* Wv = (const float*)d_in[9];  const float* bv = (const float*)d_in[10];
  const float* Ws = (const float*)d_in[11]; const float* bs = (const float*)d_in[12];
  const float* lng = (const float*)d_in[13];
  const float* lnb = (const float*)d_in[14];
  const float* W1 = (const float*)d_in[15]; const float* b1 = (const float*)d_in[16];
  const float* W2 = (const float*)d_in[17]; const float* b2 = (const float*)d_in[18];
  const float* W3 = (const float*)d_in[19]; const float* b3 = (const float*)d_in[20];
  float* out = (float*)d_out;

  char* ws = (char*)d_ws;
  size_t off = 0;
  auto alloc = [&](size_t bytes) -> void* {
    void* p = ws + off;
    off += (bytes + 255) & ~(size_t)255;
    return p;
  };
  const size_t rowbuf = (size_t)N_ENT * H_DIM * sizeof(float);   // 64 MB
  float* q     = (float*)alloc(rowbuf);
  float* k     = (float*)alloc(rowbuf);
  float* v     = (float*)alloc(rowbuf);
  float* ssb   = (float*)alloc(rowbuf);                           // x@Ws + bs
  float* msg   = (float*)alloc(rowbuf);                           // msg -> h (in place)
  float* msgsb = (float*)alloc(rowbuf);                           // per-type tx scatter
  float* h1    = (float*)alloc(rowbuf);
  float* h2    = (float*)alloc((size_t)NUM_TX * 64 * sizeof(float));
  float* sc    = (float*)alloc((size_t)E_EDGES * sizeof(float));
  float* ewb   = (float*)alloc((size_t)E_EDGES * sizeof(float));
  uint32_t* mkey = (uint32_t*)alloc((size_t)N_ENT * sizeof(uint32_t));
  float* den   = (float*)alloc((size_t)N_ENT * sizeof(float));
  // packed bf16 weight fragments (kt-major within each matrix)
  uint32_t* pQKVS  = (uint32_t*)alloc(4 * 4 * 8 * 1024);          // 4 mats, Kt=4, Nt=8
  uint32_t* pW1top = (uint32_t*)alloc(13 * 8 * 1024);             // Kt=13, Nt=8
  uint32_t* pW1typ = (uint32_t*)alloc((size_t)T_TYPES * 4 * 8 * 1024);
  uint32_t* pW2    = (uint32_t*)alloc(4 * 4 * 1024);              // Kt=4, Nt=4

  // ---- pack weights (bf16 B fragments)
  pack_b_kernel<<<4, 256, 0, stream>>>(Wq, 128, 4, 8, 128, pQKVS + 0 * 8192);
  pack_b_kernel<<<4, 256, 0, stream>>>(Wk, 128, 4, 8, 128, pQKVS + 1 * 8192);
  pack_b_kernel<<<4, 256, 0, stream>>>(Wv, 128, 4, 8, 128, pQKVS + 2 * 8192);
  pack_b_kernel<<<4, 256, 0, stream>>>(Ws, 128, 4, 8, 128, pQKVS + 3 * 8192);
  pack_b_kernel<<<13, 256, 0, stream>>>(W1, TXF, 13, 8, 128, pW1top);
  for (int t = 0; t < T_TYPES; ++t)
    pack_b_kernel<<<4, 256, 0, stream>>>(W1 + (size_t)(TXF + t * H_DIM) * H_DIM,
                                         128, 4, 8, 128, pW1typ + (size_t)t * 8192);
  pack_b_kernel<<<2, 256, 0, stream>>>(W2, 128, 4, 4, 64, pW2);

  // ---- h1 = tx_x @ W1[0:394] + b1  (K padded 394 -> 13*32)
  gemm_wmma_kernel<8><<<1024, 256, 0, stream>>>(tx_x, TXF, TXF, 13, pW1top,
                                                b1, h1, H_DIM, /*mode=*/0);

  const float inv_sqrt_d = 0.088388347648318447f;  // 1/sqrt(128)

  for (int t = 0; t < T_TYPES; ++t) {
    const float* embT = emb + (size_t)t * N_ENT * H_DIM;
    const int* idxT = eidx + (size_t)t * N_ENT;
    const int* srcT = esrc + (size_t)t * E_EDGES;
    const int* dstT = edst + (size_t)t * E_EDGES;

    hipMemsetAsync(mkey, 0, (size_t)N_ENT * 4, stream);
    hipMemsetAsync(den, 0, (size_t)N_ENT * 4, stream);
    hipMemsetAsync(msg, 0, rowbuf, stream);
    hipMemsetAsync(msgsb, 0, rowbuf, stream);

    qkvs_kernel<<<N_ENT / 16, 128, 0, stream>>>(embT, idxT, pQKVS,
                                                bq, bk, bv, bs, q, k, v, ssb);
    score_kernel<<<E_EDGES / 8, 256, 0, stream>>>(q, k, srcT, dstT, sc, mkey, inv_sqrt_d);
    expsum_kernel<<<E_EDGES / 256, 256, 0, stream>>>(sc, dstT, mkey, ewb, den);
    msg_kernel<<<E_EDGES / 8, 256, 0, stream>>>(v, ewb, den, srcT, dstT, msg);
    norm_kernel<<<N_ENT / 8, 256, 0, stream>>>(msg, ssb,
                                               lng + (size_t)t * H_DIM,
                                               lnb + (size_t)t * H_DIM);
    scatter_kernel<<<E_EDGES / 8, 256, 0, stream>>>(msg, srcT, dstT, msgsb);
    // h1 += msgs_t @ W1_t
    gemm_wmma_kernel<8><<<1024, 256, 0, stream>>>(msgsb, H_DIM, H_DIM, 4,
                                                  pW1typ + (size_t)t * 8192,
                                                  nullptr, h1, H_DIM, /*mode=*/1);
  }

  // h1 = relu(h1)
  relu_kernel<<<(NUM_TX * H_DIM / 4) / 256, 256, 0, stream>>>((float4*)h1);
  // h2 = relu(h1 @ W2 + b2)
  gemm_wmma_kernel<4><<<1024, 256, 0, stream>>>(h1, H_DIM, H_DIM, 4, pW2,
                                                b2, h2, 64, /*mode=*/2);
  // out = h2 @ W3 + b3
  out_kernel<<<NUM_TX / 8, 256, 0, stream>>>(h2, W3, b3, out);
}
